// GIN_EASY_50388556317290
// MI455X (gfx1250) — matrix operand order repaired and checked by reading, not compiled, not run on hardware
//
#include <hip/hip_runtime.h>

// ---------------------------------------------------------------------------
// GIN forward on MI455X (gfx1250): bf16 WMMA MLPs with fused BatchNorm stats,
// atomic edge aggregation, per-graph pooling, fused readout.
// ---------------------------------------------------------------------------

typedef __bf16 v16bf __attribute__((ext_vector_type(16)));
typedef float  v8f   __attribute__((ext_vector_type(8)));

union Frag32 { uint4 q[2]; v16bf v; };
static_assert(sizeof(Frag32) == 32, "frag size");

#define NN 100000   // nodes
#define NE 3200000  // edges
#define NG 512      // graphs
#define HID 64
#define DIN 128

// ---------------- small utility kernels ----------------

__global__ void zero_kernel(float* __restrict__ p, long n) {
  long i = (long)blockIdx.x * blockDim.x + threadIdx.x;
  if (i < n) p[i] = 0.0f;
}

__global__ void f32_to_bf16_kernel(const float* __restrict__ in,
                                   __bf16* __restrict__ out, long n) {
  long i = (long)blockIdx.x * blockDim.x + threadIdx.x;
  if (i < n) out[i] = (__bf16)in[i];
}

__global__ void counts_kernel(const int* __restrict__ batch,
                              float* __restrict__ cnt, int n) {
  int i = blockIdx.x * blockDim.x + threadIdx.x;
  if (i < n) atomicAdd(&cnt[batch[i]], 1.0f);
}

// one wave per node, lane handles features {lane, lane+32}
__global__ void pool_kernel(const float* __restrict__ x,
                            const int* __restrict__ batch,
                            float* __restrict__ pooled, int n) {
  int t = blockIdx.x * blockDim.x + threadIdx.x;
  int w = t >> 5, lane = t & 31;
  if (w < n) {
    int b = batch[w];
    atomicAdd(&pooled[b * HID + lane],      x[(long)w * HID + lane]);
    atomicAdd(&pooled[b * HID + lane + 32], x[(long)w * HID + lane + 32]);
  }
}

// one wave per edge: agg[dst] += x[src]  (scatter-add, L2 atomics)
__global__ void agg_kernel(const float* __restrict__ x,
                           const int* __restrict__ src,
                           const int* __restrict__ dst,
                           float* __restrict__ agg, int ne) {
  int t = blockIdx.x * blockDim.x + threadIdx.x;
  int e = t >> 5, lane = t & 31;
  if (e < ne) {
    int s = src[e], d = dst[e];
    atomicAdd(&agg[(long)d * HID + lane],      x[(long)s * HID + lane]);
    atomicAdd(&agg[(long)d * HID + lane + 32], x[(long)s * HID + lane + 32]);
  }
}

// y = bf16((1+eps)*x + agg)
__global__ void combine_kernel(const float* __restrict__ x,
                               const float* __restrict__ agg,
                               const float* __restrict__ eps,
                               __bf16* __restrict__ out, long n) {
  long i = (long)blockIdx.x * blockDim.x + threadIdx.x;
  if (i < n) {
    float e = 1.0f + eps[0];
    out[i] = (__bf16)(e * x[i] + agg[i]);
  }
}

// scale = gamma * rsqrt(var+eps), shift = beta - mean*scale
__global__ void bn_finalize_kernel(const float* __restrict__ gsum,
                                   const float* __restrict__ gsq,
                                   const float* __restrict__ gamma,
                                   const float* __restrict__ beta,
                                   float* __restrict__ scale,
                                   float* __restrict__ shift, float invN) {
  int n = threadIdx.x;
  if (n < HID) {
    float m = gsum[n] * invN;
    float v = gsq[n] * invN - m * m;
    float sc = gamma[n] * rsqrtf(v + 1e-5f);
    scale[n] = sc;
    shift[n] = beta[n] - m * sc;
  }
}

__global__ void bnrelu_bf16_kernel(const float* __restrict__ h,
                                   const float* __restrict__ scale,
                                   const float* __restrict__ shift,
                                   __bf16* __restrict__ out, long n) {
  long i = (long)blockIdx.x * blockDim.x + threadIdx.x;
  if (i < n) {
    int c = (int)(i & 63);
    float v = h[i] * scale[c] + shift[c];
    out[i] = (__bf16)(v > 0.0f ? v : 0.0f);
  }
}

__global__ void bnrelu_f32_kernel(const float* __restrict__ h,
                                  const float* __restrict__ scale,
                                  const float* __restrict__ shift,
                                  float* __restrict__ out, long n) {
  long i = (long)blockIdx.x * blockDim.x + threadIdx.x;
  if (i < n) {
    int c = (int)(i & 63);
    float v = h[i] * scale[c] + shift[c];
    out[i] = v > 0.0f ? v : 0.0f;
  }
}

// ---------------- WMMA matmul with fused BN statistics ----------------
// H[M x 64] = A[M x K](bf16) @ W[K x 64](f32 weights, staged to LDS as bf16)
// + bias. Also accumulates per-column sum / sum-of-squares for BatchNorm.
// Block = 256 threads = 8 waves; wave owns a 16-row M tile x all 64 columns.
template <int K>
__launch_bounds__(256)
__global__ void mm_bn_kernel(const __bf16* __restrict__ A,
                             const float* __restrict__ W,
                             const float* __restrict__ bias,
                             float* __restrict__ Hout,
                             float* __restrict__ gsum,
                             float* __restrict__ gsq, int M) {
  constexpr int KT = K / 32;        // K-steps of 32 (bf16 WMMA)
  constexpr int SWS = K + 8;        // padded LDS stride (bank spread)
  __shared__ __align__(16) __bf16 sW[HID * SWS];  // sW[n*SWS + k] = W[k][n]
  __shared__ float ssum[HID];
  __shared__ float ssq[HID];

  const int tid = threadIdx.x;
  if (tid < HID) { ssum[tid] = 0.0f; ssq[tid] = 0.0f; }
  for (int i = tid; i < HID * K; i += 256) {
    int n = i / K, k = i % K;
    sW[n * SWS + k] = (__bf16)W[k * HID + n];
  }
  __syncthreads();

  const int lane = tid & 31;
  const int wave = tid >> 5;
  const int half = lane >> 4;   // K-half selector for A/B fragments
  const int lrow = lane & 15;
  const int m0 = (blockIdx.x * 8 + wave) * 16;

  // ---- load A fragments (ISA 16-bit A layout: two 16B runs per lane) ----
  int arow = m0 + lrow;
  if (arow >= M) arow = M - 1;             // clamp; garbage rows never stored
  const __bf16* aptr = A + (long)arow * K;

  Frag32 af[KT];
#pragma unroll
  for (int kt = 0; kt < KT; ++kt) {
    const int k0 = kt * 32;
    af[kt].q[0] = *(const uint4*)(aptr + k0 + half * 8);
    af[kt].q[1] = *(const uint4*)(aptr + k0 + 16 + half * 8);
  }

  v8f acc[4];
#pragma unroll
  for (int nt = 0; nt < 4; ++nt)
#pragma unroll
    for (int j = 0; j < 8; ++j) acc[nt][j] = 0.0f;

  // ---- WMMA main loop: 4 N-tiles x KT K-steps ----
#pragma unroll
  for (int nt = 0; nt < 4; ++nt) {
    const __bf16* bbase = sW + (nt * 16 + lrow) * SWS;
#pragma unroll
    for (int kt = 0; kt < KT; ++kt) {
      Frag32 bfm;
      const __bf16* bp = bbase + kt * 32 + half * 8;  // two 8-elem runs: k=8h.., 16+8h..
      bfm.q[0] = *(const uint4*)(bp);
      bfm.q[1] = *(const uint4*)(bp + 16);
      acc[nt] = __builtin_amdgcn_wmma_f32_16x16x32_bf16(
          false, af[kt].v, false, bfm.v, (short)0, acc[nt], false, false);
    }
  }

  // ---- epilogue: bias add, store, fused BN partial sums ----
#pragma unroll
  for (int nt = 0; nt < 4; ++nt) {
    const int n = nt * 16 + lrow;
    const float bn = bias[n];
    float s = 0.0f, s2 = 0.0f;
#pragma unroll
    for (int r = 0; r < 8; ++r) {
      const int row = m0 + half * 8 + r;   // D layout: VGPR r, half selects +8
      const float v = acc[nt][r] + bn;
      if (row < M) {
        Hout[(long)row * HID + n] = v;
        s += v;
        s2 += v * v;
      }
    }
    atomicAdd(&ssum[n], s);
    atomicAdd(&ssq[n], s2);
  }
  __syncthreads();
  if (tid < HID) {
    atomicAdd(&gsum[tid], ssum[tid]);
    atomicAdd(&gsq[tid], ssq[tid]);
  }
}

// ---------------- readout: out[g] = sum_l pooled_l[g]@W_l + biases ----------
__global__ void final_kernel(const float* __restrict__ pooled,
                             const float* __restrict__ counts,
                             const float* W0, const float* b0,
                             const float* W1, const float* b1,
                             const float* W2, const float* b2,
                             const float* W3, const float* b3,
                             const float* W4, const float* b4,
                             float* __restrict__ out) {
  int g = blockIdx.x;
  int j = threadIdx.x;
  if (g >= NG || j >= 10) return;
  const float* Ws[5] = {W0, W1, W2, W3, W4};
  const float* bs[5] = {b0, b1, b2, b3, b4};
  float acc = counts[g] * b0[j];  // layer-0 bias summed per node
  for (int l = 0; l < 5; ++l) {
    const float* p = pooled + ((long)l * NG + g) * HID;
    const float* Wl = Ws[l];
    float a = 0.0f;
    for (int k = 0; k < HID; ++k) a += p[k] * Wl[k * 10 + j];
    acc += a;
    if (l > 0) acc += bs[l][j];  // later biases added once per graph
  }
  out[g * 10 + j] = acc;
}

// ---------------------------------------------------------------------------

extern "C" void kernel_launch(void* const* d_in, const int* in_sizes, int n_in,
                              void* d_out, int out_size, void* d_ws,
                              size_t ws_size, hipStream_t stream) {
  (void)in_sizes; (void)n_in; (void)out_size; (void)ws_size;

  // ---- inputs (pytree leaf order of setup_inputs) ----
  const float* x_in  = (const float*)d_in[0];
  const int*   ei    = (const int*)d_in[1];
  const int*   batch = (const int*)d_in[2];
  const int* src = ei;
  const int* dst = ei + NE;
  // first_h_params: W1,b1,g1,be1,W2,b2,g2,be2 at d_in[3..10]
  // conv l params : eps,W1,b1,g1,be1,W2,b2,g2,be2 at d_in[11+9l ..]
  // linear l      : W,b at d_in[47+2l], d_in[48+2l]

  // ---- workspace carve-up ----
  size_t off = 0;
  auto take = [&](size_t bytes) -> void* {
    void* p = (char*)d_ws + off;
    off = (off + bytes + 255) & ~(size_t)255;
    return p;
  };
  __bf16* XB   = (__bf16*)take((size_t)NN * DIN * 2);  // bf16 input features
  __bf16* AB   = (__bf16*)take((size_t)NN * HID * 2);  // bf16 activations
  float*  Hbuf = (float*)take((size_t)NN * HID * 4);   // pre-BN matmul out
  float*  X    = (float*)take((size_t)NN * HID * 4);   // current features f32
  float*  AGG  = (float*)take((size_t)NN * HID * 4);   // neighbor sums
  float*  POOL = (float*)take((size_t)5 * NG * HID * 4);
  float*  CNT  = (float*)take((size_t)NG * 4);
  float*  SUM  = (float*)take((size_t)2 * HID * 4);    // sum | sumsq
  float*  SQ   = SUM + HID;
  float*  SCALE = (float*)take((size_t)HID * 4);
  float*  SHIFT = (float*)take((size_t)HID * 4);

  const long nElem = (long)NN * HID;
  const int eBlocks = (int)((nElem + 255) / 256);
  const int mmBlocks = (NN + 127) / 128;  // 8 waves x 16 rows per block

  auto zero = [&](float* p, long n) {
    zero_kernel<<<(int)((n + 255) / 256), 256, 0, stream>>>(p, n);
  };
  auto bnfin = [&](const float* g, const float* be) {
    bn_finalize_kernel<<<1, 64, 0, stream>>>(SUM, SQ, g, be, SCALE, SHIFT,
                                             1.0f / (float)NN);
  };

  // ---- init per-call state ----
  zero(POOL, (long)5 * NG * HID);
  zero(CNT, NG);
  counts_kernel<<<(NN + 255) / 256, 256, 0, stream>>>(batch, CNT, NN);

  // ---- first MLP: 128 -> 64 -> 64 ----
  f32_to_bf16_kernel<<<(int)(((long)NN * DIN + 255) / 256), 256, 0, stream>>>(
      x_in, XB, (long)NN * DIN);
  zero(SUM, 2 * HID);
  mm_bn_kernel<DIN><<<mmBlocks, 256, 0, stream>>>(
      XB, (const float*)d_in[3], (const float*)d_in[4], Hbuf, SUM, SQ, NN);
  bnfin((const float*)d_in[5], (const float*)d_in[6]);
  bnrelu_bf16_kernel<<<eBlocks, 256, 0, stream>>>(Hbuf, SCALE, SHIFT, AB, nElem);
  zero(SUM, 2 * HID);
  mm_bn_kernel<HID><<<mmBlocks, 256, 0, stream>>>(
      AB, (const float*)d_in[7], (const float*)d_in[8], Hbuf, SUM, SQ, NN);
  bnfin((const float*)d_in[9], (const float*)d_in[10]);
  bnrelu_f32_kernel<<<eBlocks, 256, 0, stream>>>(Hbuf, SCALE, SHIFT, X, nElem);

  // layer-0 pooling
  pool_kernel<<<(NN * 32 + 255) / 256, 256, 0, stream>>>(X, batch, POOL, NN);

  // ---- 4 GIN convs ----
  for (int l = 0; l < 4; ++l) {
    const int pb = 11 + l * 9;  // eps, W1,b1,g1,be1, W2,b2,g2,be2
    zero(AGG, nElem);
    agg_kernel<<<(int)(((long)NE * 32 + 255) / 256), 256, 0, stream>>>(
        X, src, dst, AGG, NE);
    combine_kernel<<<eBlocks, 256, 0, stream>>>(
        X, AGG, (const float*)d_in[pb + 0], AB, nElem);

    zero(SUM, 2 * HID);
    mm_bn_kernel<HID><<<mmBlocks, 256, 0, stream>>>(
        AB, (const float*)d_in[pb + 1], (const float*)d_in[pb + 2], Hbuf, SUM,
        SQ, NN);
    bnfin((const float*)d_in[pb + 3], (const float*)d_in[pb + 4]);
    bnrelu_bf16_kernel<<<eBlocks, 256, 0, stream>>>(Hbuf, SCALE, SHIFT, AB,
                                                    nElem);
    zero(SUM, 2 * HID);
    mm_bn_kernel<HID><<<mmBlocks, 256, 0, stream>>>(
        AB, (const float*)d_in[pb + 5], (const float*)d_in[pb + 6], Hbuf, SUM,
        SQ, NN);
    bnfin((const float*)d_in[pb + 7], (const float*)d_in[pb + 8]);
    bnrelu_f32_kernel<<<eBlocks, 256, 0, stream>>>(Hbuf, SCALE, SHIFT, X,
                                                   nElem);

    pool_kernel<<<(NN * 32 + 255) / 256, 256, 0, stream>>>(
        X, batch, POOL + (long)(l + 1) * NG * HID, NN);
  }

  // ---- fused readout ----
  final_kernel<<<NG, 16, 0, stream>>>(
      POOL, CNT,
      (const float*)d_in[47], (const float*)d_in[48],
      (const float*)d_in[49], (const float*)d_in[50],
      (const float*)d_in[51], (const float*)d_in[52],
      (const float*)d_in[53], (const float*)d_in[54],
      (const float*)d_in[55], (const float*)d_in[56],
      (float*)d_out);
}
